// DHILModel_44779329028372
// MI455X (gfx1250) — matrix operand order repaired and checked by reading, not compiled, not run on hardware
//
#include <hip/hip_runtime.h>
#include <math.h>
#include <stddef.h>

// ---------------------------------------------------------------------------
// MI455X (gfx1250) implementation of the two-level GNN reference.
//  - All dense matmuls use v_wmma_f32_16x16x32_bf16 (fp32 -> bf16, f32 accum).
//  - N-tile count is a template parameter so accumulators fully unroll into
//    VGPRs (round-1 version spilled acc[] to scratch due to runtime NT).
//  - Bipartite softmax attention fused flash-style (stats pass + on-the-fly
//    exp weights as bf16 A-fragments); attention matrix never materialized.
// ---------------------------------------------------------------------------

typedef __attribute__((ext_vector_type(16))) __bf16 v16bf;
typedef __attribute__((ext_vector_type(8)))  float  v8f;

#define NEG_SLOPE 0.2f

union FragAB {
  v16bf v;
  unsigned u[8];
  __bf16 e[16];
};

static __device__ __forceinline__ float lrelu_(float x) { return x > 0.f ? x : NEG_SLOPE * x; }
static __device__ __forceinline__ float sigm_(float x) { return 1.f / (1.f + __expf(-x)); }

// float -> bf16 round-to-nearest-even via bit ops
static __device__ __forceinline__ __bf16 f2bf(float f) {
  unsigned u = __builtin_bit_cast(unsigned, f);
  unsigned short h = (unsigned short)((u + 0x7FFFu + ((u >> 16) & 1u)) >> 16);
  return __builtin_bit_cast(__bf16, h);
}

// ---------------------------------------------------------------------------
// WMMA GEMM: Cout[M,N] = epil(A[M,K] @ W[K,N] (+ Cin)),  N = NT*16 (template).
// A,W,Cin,Cout f32 row-major. M%64==0, K%32==0.
// epil: 0 none, 1 leaky-relu(0.2), 2 sigmoid.
// 128 threads (4 waves); each wave computes a 16xN strip.
// ---------------------------------------------------------------------------
template <int NT>
__global__ __launch_bounds__(128) void gemm_bf16_kernel(
    const float* __restrict__ A, const float* __restrict__ W,
    const float* __restrict__ Cin, float* __restrict__ Cout,
    int M, int K, int epil)
{
  constexpr int N = NT * 16;
  __shared__ __attribute__((aligned(16))) __bf16 sA[64 * 32];  // [m][k]
  __shared__ __attribute__((aligned(16))) __bf16 sW[N * 32];   // [n][k] K-contiguous
  const int tid  = threadIdx.x;
  const int lane = tid & 31;
  const int wid  = tid >> 5;
  const int m0   = blockIdx.x * 64;
  const int half8  = (lane >> 4) << 3;   // 0 or 8
  const int half16 = (lane >> 4) << 4;   // 0 or 16

  v8f zero = {};
  v8f acc[NT];
#pragma unroll
  for (int i = 0; i < NT; ++i) acc[i] = zero;

  const int mloc = (wid << 4) + (lane & 15);

  for (int kk = 0; kk < K; kk += 32) {
#pragma unroll
    for (int idx = tid, it = 0; it < 16; ++it, idx += 128) {
      int r = idx >> 5, c = idx & 31;
      sA[idx] = f2bf(A[(size_t)(m0 + r) * K + kk + c]);
    }
#pragma unroll
    for (int idx = tid, it = 0; it < (32 * N) / 128; ++it, idx += 128) {
      int k = idx / N, n = idx - k * N;
      sW[n * 32 + k] = f2bf(W[(size_t)(kk + k) * N + n]);
    }
    if (kk + 32 < K) {  // global_prefetch_b8 for next tiles
      __builtin_prefetch(&A[(size_t)(m0 + (tid >> 1)) * K + kk + 32]);
      __builtin_prefetch(&W[(size_t)(kk + 32 + lane) * N]);
    }
    __syncthreads();

    // A fragment (16x32 bf16): lane<16 K=0..15, lane>=16 K=16..31 (matches B)
    FragAB aF;
    const __bf16* arow = &sA[mloc * 32];
#pragma unroll
    for (int p = 0; p < 8; ++p) {
      int k = (p << 1) + half16;
      aF.u[p] = *(const unsigned*)(arow + k);
    }
#pragma unroll
    for (int nt = 0; nt < NT; ++nt) {
      FragAB bF;
      const __bf16* wrow = &sW[(nt * 16 + (lane & 15)) * 32 + half16];
#pragma unroll
      for (int p = 0; p < 8; ++p) bF.u[p] = *(const unsigned*)(wrow + (p << 1));
      acc[nt] = __builtin_amdgcn_wmma_f32_16x16x32_bf16(
          false, aF.v, false, bF.v, (short)0, acc[nt], false, false);
    }
    __syncthreads();
  }

  const int rbase = m0 + (wid << 4) + half8;
  const int cbase = lane & 15;
#pragma unroll
  for (int nt = 0; nt < NT; ++nt) {
#pragma unroll
    for (int r = 0; r < 8; ++r) {
      int row = rbase + r;
      int col = (nt << 4) + cbase;
      float v = acc[nt][r];
      if (Cin) v += Cin[(size_t)row * N + col];
      if (epil == 1) v = lrelu_(v);
      else if (epil == 2) v = sigm_(v);
      Cout[(size_t)row * N + col] = v;
    }
  }
}

// ---------------------------------------------------------------------------
// Per-row softmax stats over the opposite node set: m = max_j(-d2), s = sum exp
// One wave per query row.
// ---------------------------------------------------------------------------
__global__ __launch_bounds__(256) void softmax_stats_kernel(
    const float* __restrict__ Qc, const float* __restrict__ Kc,
    float* __restrict__ mOut, float* __restrict__ sOut, int M, int J)
{
  const int lane = threadIdx.x & 31;
  const int row  = blockIdx.x * 8 + (threadIdx.x >> 5);
  const float qx = Qc[row * 3 + 0], qy = Qc[row * 3 + 1], qz = Qc[row * 3 + 2];
  float m = -3.402823466e38f, s = 0.f;
  for (int j = lane; j < J; j += 32) {
    float dx = qx - Kc[j * 3 + 0], dy = qy - Kc[j * 3 + 1], dz = qz - Kc[j * 3 + 2];
    float x = -(dx * dx + dy * dy + dz * dz);
    float nm = fmaxf(m, x);
    s = s * __expf(m - nm) + __expf(x - nm);
    m = nm;
  }
  for (int off = 16; off > 0; off >>= 1) {
    float om = __shfl_xor(m, off, 32);
    float os = __shfl_xor(s, off, 32);
    float nm = fmaxf(m, om);
    s = s * __expf(m - nm) + os * __expf(om - nm);
    m = nm;
  }
  if (lane == 0) { mOut[row] = m; sOut[row] = s; }
}

// ---------------------------------------------------------------------------
// Fused attention apply: Out[i,:] = (1/s_i) * sum_j exp(-d2(i,j)-m_i) * Z[j,:]
// D = NT*16 (template). M%64==0, J%32==0.
// ---------------------------------------------------------------------------
template <int NT>
__global__ __launch_bounds__(128) void attn_apply_kernel(
    const float* __restrict__ Qc, const float* __restrict__ Kc,
    const float* __restrict__ mRow, const float* __restrict__ sRow,
    const float* __restrict__ Z, float* __restrict__ Out,
    int M, int J)
{
  constexpr int D = NT * 16;
  __shared__ __attribute__((aligned(16))) __bf16 sZ[D * 32];  // [d][j] K-contig
  __shared__ float sKc[32 * 3];
  const int tid  = threadIdx.x;
  const int lane = tid & 31;
  const int wid  = tid >> 5;
  const int m0   = blockIdx.x * 64;
  const int half8  = (lane >> 4) << 3;
  const int half16 = (lane >> 4) << 4;
  const int myRow = m0 + (wid << 4) + (lane & 15);
  const float qx = Qc[myRow * 3 + 0], qy = Qc[myRow * 3 + 1], qz = Qc[myRow * 3 + 2];
  const float mi = mRow[myRow];

  v8f zero = {};
  v8f acc[NT];
#pragma unroll
  for (int i = 0; i < NT; ++i) acc[i] = zero;

  for (int j0 = 0; j0 < J; j0 += 32) {
#pragma unroll
    for (int idx = tid, it = 0; it < (32 * D) / 128; ++it, idx += 128) {
      int jj = idx / D, d = idx - jj * D;
      sZ[d * 32 + jj] = f2bf(Z[(size_t)(j0 + jj) * D + d]);
    }
    if (tid < 96) sKc[tid] = Kc[j0 * 3 + tid];
    if (j0 + 32 < J) __builtin_prefetch(&Z[(size_t)(j0 + 32 + (tid >> 2)) * D]);
    __syncthreads();

    FragAB aF;  // on-the-fly softmax weights, A layout
#pragma unroll
    for (int i = 0; i < 16; ++i) {
      int k = i + half16;
      float dx = qx - sKc[k * 3 + 0];
      float dy = qy - sKc[k * 3 + 1];
      float dz = qz - sKc[k * 3 + 2];
      float d2 = dx * dx + dy * dy + dz * dz;
      aF.e[i] = f2bf(__expf(-d2 - mi));
    }
#pragma unroll
    for (int nt = 0; nt < NT; ++nt) {
      FragAB bF;
      const __bf16* zr = &sZ[(nt * 16 + (lane & 15)) * 32 + half16];
#pragma unroll
      for (int p = 0; p < 8; ++p) bF.u[p] = *(const unsigned*)(zr + (p << 1));
      acc[nt] = __builtin_amdgcn_wmma_f32_16x16x32_bf16(
          false, aF.v, false, bF.v, (short)0, acc[nt], false, false);
    }
    __syncthreads();
  }

  const int rbase = m0 + (wid << 4) + half8;
  const int cbase = lane & 15;
#pragma unroll
  for (int nt = 0; nt < NT; ++nt) {
#pragma unroll
    for (int r = 0; r < 8; ++r) {
      int row = rbase + r;
      float inv = 1.f / sRow[row];
      Out[(size_t)row * D + (nt << 4) + cbase] = acc[nt][r] * inv;
    }
  }
}

// --------------------------- elementwise / sparse ---------------------------

__global__ void fill_kernel(float* __restrict__ p, float v, int n) {
  int t = blockIdx.x * blockDim.x + threadIdx.x;
  if (t < n) p[t] = v;
}

__global__ void scatter_edges_kernel(const float* __restrict__ X,
                                     const int* __restrict__ src,
                                     const int* __restrict__ dst,
                                     float* __restrict__ acc, int total, int D) {
  int t = blockIdx.x * blockDim.x + threadIdx.x;
  if (t >= total) return;
  int e = t / D, d = t - e * D;
  atomicAdd(&acc[(size_t)dst[e] * D + d], X[(size_t)src[e] * D + d]);
}

__global__ void combine_lrelu_kernel(const float* __restrict__ acc,
                                     const float* __restrict__ hs,
                                     const float* __restrict__ b,
                                     float* __restrict__ out, int total, int N) {
  int t = blockIdx.x * blockDim.x + threadIdx.x;
  if (t >= total) return;
  int c = t % N;
  out[t] = lrelu_(acc[t] + hs[t] + b[c]);
}

__global__ void gate_apply_kernel(float* __restrict__ intra,
                                  float* __restrict__ inter,
                                  const float* __restrict__ g,
                                  const int* __restrict__ mask,
                                  int total, int N) {
  int t = blockIdx.x * blockDim.x + threadIdx.x;
  if (t >= total) return;
  int r = t / N;
  float gi = g[t] * inter[t];
  inter[t] = gi;
  float mk = mask ? (float)mask[r] : 1.f;
  intra[t] += mk * gi;
}

__global__ void sub_init_kernel(const float* __restrict__ x,
                                float* __restrict__ out, int total, int DS, int SUBX) {
  int t = blockIdx.x * blockDim.x + threadIdx.x;
  if (t >= total) return;
  int r = t / DS, c = t - r * DS;
  out[t] = (c < SUBX) ? x[(size_t)r * SUBX + c] : 0.f;
}

__global__ void scatter_feat_kernel(const float* __restrict__ H,
                                    const int* __restrict__ map,
                                    float* __restrict__ out,
                                    int total, int D, int DS, int off) {
  int t = blockIdx.x * blockDim.x + threadIdx.x;
  if (t >= total) return;
  int a = t / D, d = t - a * D;
  atomicAdd(&out[(size_t)map[a] * DS + off + d], H[t]);
}

__global__ __launch_bounds__(256) void colmean_kernel(
    const float* __restrict__ X, int M, int N, float* __restrict__ out) {
  __shared__ float red[256];
  const int c = blockIdx.x;
  float s = 0.f;
  for (int r = threadIdx.x; r < M; r += 256) s += X[(size_t)r * N + c];
  red[threadIdx.x] = s;
  __syncthreads();
  for (int st = 128; st > 0; st >>= 1) {
    if (threadIdx.x < st) red[threadIdx.x] += red[threadIdx.x + st];
    __syncthreads();
  }
  if (threadIdx.x == 0) out[c] = red[0] / (float)M;
}

// --------------------------- GRU head (tiny) --------------------------------
__global__ __launch_bounds__(384) void gru_head_kernel(
    const float* __restrict__ Wih, const float* __restrict__ Whh,
    const float* __restrict__ bih, const float* __restrict__ bhh,
    const float* __restrict__ Wp, const float* __restrict__ bp,
    const float* __restrict__ Hf, const float* __restrict__ Zf,
    float* __restrict__ out) {
  __shared__ float h[384];
  __shared__ float red[512];
  const int j = threadIdx.x;
  h[j] = 0.f;
  if (j < 128) red[384 + j] = 0.f;
  __syncthreads();
  for (int t = 0; t < 2; ++t) {
    const float* x = (t == 0) ? Hf : Zf;
    float ir = bih[j], iz = bih[384 + j], in = bih[768 + j];
    float hr = bhh[j], hz = bhh[384 + j], hn = bhh[768 + j];
    for (int k = 0; k < 384; ++k) {
      float xv = x[k], hv = h[k];
      const float* wi = &Wih[(size_t)k * 1152];
      const float* wh = &Whh[(size_t)k * 1152];
      ir += xv * wi[j];       iz += xv * wi[384 + j];  in += xv * wi[768 + j];
      hr += hv * wh[j];       hz += hv * wh[384 + j];  hn += hv * wh[768 + j];
    }
    float r = sigm_(ir + hr);
    float z = sigm_(iz + hz);
    float n = tanhf(in + r * hn);
    float hnew = (1.f - z) * n + z * h[j];
    __syncthreads();
    h[j] = hnew;
    __syncthreads();
  }
  red[j] = h[j] * Wp[j];
  __syncthreads();
  for (int st = 256; st > 0; st >>= 1) {
    if (j < st) red[j] += red[j + st];
    __syncthreads();
  }
  if (j == 0) out[0] = red[0] + bp[0];
}

// ---------------------------------------------------------------------------
// Host driver
// ---------------------------------------------------------------------------
#define CEILDIV(a, b) (((a) + (b) - 1) / (b))

static const int D_ATOM = 128, SUBX = 64, DS_SUB = 192;
static const int N_LIG = 2048, N_PROT = 16384, N_FRAG = 256, N_RES = 2048;
static const int E_LIG = 16384, E_PROT = 262144, E_FRAG = 2048, E_RES = 32768;

// d_in index map.  Assumption: `params` (dict) is pytree-flattened with
// alphabetical keys / lists in order, followed by the remaining setup_inputs()
// entries in insertion order.
enum {
  IX_Wp = 0, IX_bp = 1,
  IX_frag_intra = 2,   // 3 x {Wm, Ws, b}
  IX_gru_Whh = 11, IX_gru_Wih = 12, IX_gru_bhh = 13, IX_gru_bih = 14,
  IX_inter_atom = 15,  // 2 x {Wl, Wp}
  IX_inter_sub = 19,   // 2 x {Wl, Wp}
  IX_lig_act = 23,     // 2 x Wg
  IX_lig_intra = 25,   // 3 x {Wm, Ws, b}
  IX_lig_sub_act = 34, // 2 x Wg
  IX_prot_act = 36,    // 2 x Wg
  IX_prot_intra = 38,  // 3 x {Wm, Ws, b}
  IX_prot_sub_act = 47,// 2 x Wg
  IX_res_intra = 49,   // 3 x {Wm, Ws, b}
  IX_lig_h = 58, IX_prot_h = 59, IX_lig_coord = 60, IX_prot_coord = 61,
  IX_frag_x = 62, IX_res_x = 63, IX_frag_coord = 64, IX_res_coord = 65,
  IX_lig_src = 66, IX_lig_dst = 67, IX_prot_src = 68, IX_prot_dst = 69,
  IX_lig_group = 70, IX_prot_group = 71, IX_atom2frag = 72, IX_atom2res = 73,
  IX_frag_src = 74, IX_frag_dst = 75, IX_res_src = 76, IX_res_dst = 77
};

extern "C" void kernel_launch(void* const* d_in, const int* in_sizes, int n_in,
                              void* d_out, int out_size, void* d_ws, size_t ws_size,
                              hipStream_t stream) {
  (void)in_sizes; (void)n_in; (void)out_size; (void)ws_size;
  auto F = [&](int i) { return (const float*)d_in[i]; };
  auto I = [&](int i) { return (const int*)d_in[i]; };

  size_t off = 0;
  auto alloc = [&](size_t nfloats) {
    float* p = (float*)((char*)d_ws + off);
    off += ((nfloats * sizeof(float) + 255) / 256) * 256;
    return p;
  };

  const size_t LN = (size_t)N_LIG * D_ATOM;
  float* hPingL = alloc(LN); float* hPongL = alloc(LN);
  float* scrL = alloc(LN);   float* accL = alloc(LN);
  float* hsL = alloc(LN);    float* zlB1 = alloc(LN);
  const size_t PN = (size_t)N_PROT * D_ATOM;
  float* hPingP = alloc(PN); float* hPongP = alloc(PN);
  float* scrP = alloc(PN);   float* accP = alloc(PN);
  float* hsP = alloc(PN);    float* zpB1 = alloc(PN);
  float* mL = alloc(N_LIG);  float* sL = alloc(N_LIG);
  float* mP = alloc(N_PROT); float* sP = alloc(N_PROT);
  const size_t FN = (size_t)N_FRAG * DS_SUB;
  const size_t RN = (size_t)N_RES * DS_SUB;
  float* fragH = alloc(FN);  float* resH = alloc(RN);
  float* fPing = alloc(FN);  float* fPong = alloc(FN);
  float* scrF = alloc(FN);   float* accF = alloc(FN);
  float* hsF = alloc(FN);    float* zfB1 = alloc(FN);
  float* rPing = alloc(RN);  float* rPong = alloc(RN);
  float* scrR = alloc(RN);   float* accR = alloc(RN);
  float* hsR = alloc(RN);    float* zrB1 = alloc(RN);
  float* mF = alloc(N_FRAG); float* sF = alloc(N_FRAG);
  float* mR = alloc(N_RES);  float* sR = alloc(N_RES);
  float* Hf = alloc(384);    float* Zf = alloc(384);

  auto gemm = [&](const float* A, const float* W, const float* Cin,
                  float* Cout, int M, int K, int N, int epil) {
    if (N == 128)
      gemm_bf16_kernel<8><<<M / 64, 128, 0, stream>>>(A, W, Cin, Cout, M, K, epil);
    else
      gemm_bf16_kernel<12><<<M / 64, 128, 0, stream>>>(A, W, Cin, Cout, M, K, epil);
  };
  auto attn = [&](const float* Qc, const float* Kc, const float* mRow,
                  const float* sRow, const float* Z, float* Out,
                  int M, int J, int D) {
    if (D == 128)
      attn_apply_kernel<8><<<M / 64, 128, 0, stream>>>(Qc, Kc, mRow, sRow, Z, Out, M, J);
    else
      attn_apply_kernel<12><<<M / 64, 128, 0, stream>>>(Qc, Kc, mRow, sRow, Z, Out, M, J);
  };

  // 3-layer GCN: h' = lrelu(segsum(h@Wm) + h@Ws + b); result lands in ping.
  auto gcn3 = [&](const float* hin, float* ping, float* pong, float* scr,
                  float* accb, float* hsb, int M, int D,
                  const int* src, const int* dst, int E, int wBase) {
    const float* cur = hin;
    float* outs[3] = {ping, pong, ping};
    for (int l = 0; l < 3; ++l) {
      gemm(cur, F(wBase + 3 * l + 0), nullptr, scr, M, D, D, 0);  // h@Wm
      gemm(cur, F(wBase + 3 * l + 1), nullptr, hsb, M, D, D, 0);  // h@Ws
      int total = M * D;
      fill_kernel<<<CEILDIV(total, 256), 256, 0, stream>>>(accb, 0.f, total);
      int et = E * D;
      scatter_edges_kernel<<<CEILDIV(et, 256), 256, 0, stream>>>(scr, src, dst, accb, et, D);
      combine_lrelu_kernel<<<CEILDIV(total, 256), 256, 0, stream>>>(
          accb, hsb, F(wBase + 3 * l + 2), outs[l], total, D);
      cur = outs[l];
    }
  };

  // 2-layer bipartite inter channel; final outputs in zlB/zpB.
  auto run_inter = [&](const float* cL, const float* cP,
                       const float* zlIn, const float* zpIn,
                       int ML, int MP, int D, int wBase,
                       float* mQl, float* sQl, float* mQp, float* sQp,
                       float* tmpL, float* tmpP,
                       float* zlA, float* zpA, float* zlB, float* zpB) {
    softmax_stats_kernel<<<ML / 8, 256, 0, stream>>>(cL, cP, mQl, sQl, ML, MP);
    softmax_stats_kernel<<<MP / 8, 256, 0, stream>>>(cP, cL, mQp, sQp, MP, ML);
    attn(cL, cP, mQl, sQl, zpIn, tmpL, ML, MP, D);
    gemm(tmpL, F(wBase + 0), nullptr, zlA, ML, D, D, 1);
    attn(cP, cL, mQp, sQp, zlIn, tmpP, MP, ML, D);
    gemm(tmpP, F(wBase + 1), nullptr, zpA, MP, D, D, 1);
    attn(cL, cP, mQl, sQl, zpA, tmpL, ML, MP, D);
    gemm(tmpL, F(wBase + 2), nullptr, zlB, ML, D, D, 1);
    attn(cP, cL, mQp, sQp, zlA, tmpP, MP, ML, D);
    gemm(tmpP, F(wBase + 3), nullptr, zpB, MP, D, D, 1);
  };

  // gated interactive exchange: g = sigmoid([intra,inter]@Wg); inter*=g;
  // intra += mask*inter.  concat handled as two accumulating GEMMs.
  auto interact = [&](float* intra, float* inter, float* gbuf,
                      int M, int D, int wBase, int L, const int* mask) {
    for (int l = 0; l < L; ++l) {
      const float* Wg = F(wBase + l);                    // [2D, D]
      gemm(intra, Wg, nullptr, gbuf, M, D, D, 0);
      gemm(inter, Wg + (size_t)D * D, gbuf, gbuf, M, D, D, 2);
      int total = M * D;
      gate_apply_kernel<<<CEILDIV(total, 256), 256, 0, stream>>>(
          intra, inter, gbuf, mask, total, D);
    }
  };

  // ---------------- atom level ----------------
  gcn3(F(IX_lig_h), hPingL, hPongL, scrL, accL, hsL,
       N_LIG, D_ATOM, I(IX_lig_src), I(IX_lig_dst), E_LIG, IX_lig_intra);
  gcn3(F(IX_prot_h), hPingP, hPongP, scrP, accP, hsP,
       N_PROT, D_ATOM, I(IX_prot_src), I(IX_prot_dst), E_PROT, IX_prot_intra);

  run_inter(F(IX_lig_coord), F(IX_prot_coord), hPingL, hPingP,
            N_LIG, N_PROT, D_ATOM, IX_inter_atom,
            mL, sL, mP, sP, scrL, scrP,
            hPongL, hPongP, zlB1, zpB1);

  interact(hPingL, zlB1, accL, N_LIG, D_ATOM, IX_lig_act, 2, I(IX_lig_group));
  interact(hPingP, zpB1, accP, N_PROT, D_ATOM, IX_prot_act, 2, I(IX_prot_group));
  // H_lig = hPingL, H_prot = hPingP

  // ---------------- atoms -> substructures ----------------
  {
    int tf = N_FRAG * DS_SUB;
    sub_init_kernel<<<CEILDIV(tf, 256), 256, 0, stream>>>(F(IX_frag_x), fragH, tf, DS_SUB, SUBX);
    int ta = N_LIG * D_ATOM;
    scatter_feat_kernel<<<CEILDIV(ta, 256), 256, 0, stream>>>(
        hPingL, I(IX_atom2frag), fragH, ta, D_ATOM, DS_SUB, SUBX);
    int tr = N_RES * DS_SUB;
    sub_init_kernel<<<CEILDIV(tr, 256), 256, 0, stream>>>(F(IX_res_x), resH, tr, DS_SUB, SUBX);
    int tp = N_PROT * D_ATOM;
    scatter_feat_kernel<<<CEILDIV(tp, 256), 256, 0, stream>>>(
        hPingP, I(IX_atom2res), resH, tp, D_ATOM, DS_SUB, SUBX);
  }

  // ---------------- substructure level ----------------
  gcn3(fragH, fPing, fPong, scrF, accF, hsF,
       N_FRAG, DS_SUB, I(IX_frag_src), I(IX_frag_dst), E_FRAG, IX_frag_intra);
  gcn3(resH, rPing, rPong, scrR, accR, hsR,
       N_RES, DS_SUB, I(IX_res_src), I(IX_res_dst), E_RES, IX_res_intra);

  run_inter(F(IX_frag_coord), F(IX_res_coord), fPing, rPing,
            N_FRAG, N_RES, DS_SUB, IX_inter_sub,
            mF, sF, mR, sR, scrF, scrR,
            fPong, rPong, zfB1, zrB1);

  interact(fPing, zfB1, accF, N_FRAG, DS_SUB, IX_lig_sub_act, 2, nullptr);
  interact(rPing, zrB1, accR, N_RES, DS_SUB, IX_prot_sub_act, 2, nullptr);
  // li_intra=fPing, li_inter=zfB1, pi_intra=rPing, pi_inter=zrB1

  // ---------------- pooling + GRU head ----------------
  colmean_kernel<<<DS_SUB, 256, 0, stream>>>(fPing, N_FRAG, DS_SUB, Hf);
  colmean_kernel<<<DS_SUB, 256, 0, stream>>>(rPing, N_RES, DS_SUB, Hf + DS_SUB);
  colmean_kernel<<<DS_SUB, 256, 0, stream>>>(zfB1, N_FRAG, DS_SUB, Zf);
  colmean_kernel<<<DS_SUB, 256, 0, stream>>>(zrB1, N_RES, DS_SUB, Zf + DS_SUB);

  gru_head_kernel<<<1, 384, 0, stream>>>(
      F(IX_gru_Wih), F(IX_gru_Whh), F(IX_gru_bih), F(IX_gru_bhh),
      F(IX_Wp), F(IX_bp), Hf, Zf, (float*)d_out);
}
